// Cp_linear_4D_6760278524111
// MI455X (gfx1250) — matrix operand order repaired and checked by reading, not compiled, not run on hardware
//
#include <hip/hip_runtime.h>

// Problem constants (from reference)
constexpr int kH = 32, kW = 32, kC = 512, kCLS = 1000, kR = 64, kB = 128;
constexpr int kP = kW * kH;          // 1024 spatial positions, contiguous in x
constexpr int kM = kB * kC;          // 65536 GEMM rows

typedef __attribute__((ext_vector_type(16))) __bf16 v16bf;
typedef __attribute__((ext_vector_type(8)))  __bf16 v8bf;
typedef __attribute__((ext_vector_type(8)))  float  v8f;
typedef __attribute__((ext_vector_type(4)))  float  v4f;

// ---------------------------------------------------------------------------
// Kernel 0: KT[r][p] = U1[h,r] * U2[w,r]  (bf16, r-major so staging reads are
// per-column contiguous), and zero the t3 accumulator (ws is poisoned).
// ---------------------------------------------------------------------------
__global__ void __launch_bounds__(256)
cp_build_kt(const float* __restrict__ U1, const float* __restrict__ U2,
            __bf16* __restrict__ KT, float* __restrict__ t3) {
  int idx = blockIdx.x * 256 + threadIdx.x;      // 0 .. 65535
  int r = idx >> 10;
  int p = idx & (kP - 1);
  int w = p >> 5;
  int h = p & 31;
  float v = U1[h * kR + r] * U2[w * kR + r];
  KT[(size_t)r * kP + p] = (__bf16)v;
  if (idx < kB * kR) t3[idx] = 0.0f;
}

// ---------------------------------------------------------------------------
// Kernel 1: t2[(b,c),r] = sum_p x[(b,c),p] * K[p,r]  (bf16 WMMA, f32 accum)
// fused epilogue: t3[b,r] += sum_{c in tile} t2 * U3[c,r]  (f32 atomics).
// One wave = 16 rows x 64 cols; 8 waves/block, 512 blocks = 65536 rows.
// B-matrix (KT) is staged in LDS pre-swizzled to the exact per-lane WMMA
// fragment layout: chunk(kb, j, lane) = 16 contiguous bf16 (32 B), so the
// inner loop's B reads are two ds_load_b128 per tile (no VMEM latency chain).
// ---------------------------------------------------------------------------
__global__ void __launch_bounds__(256)
cp_contract(const float* __restrict__ x, const __bf16* __restrict__ KT,
            const float* __restrict__ U3, float* __restrict__ t3) {
  // 2048 chunks x 16 bf16 = 64 KB: [kb(16)][j(4)][lane(32)][16 bf16]
  __shared__ __bf16 ldsB[2048 * 16];

  const int lane  = threadIdx.x & 31;
  const int hi    = lane >> 4;          // 0 | 1  (K-half selector)
  const int lo    = lane & 15;          // row / column within tile
  const int gwave = (blockIdx.x * 256 + (int)threadIdx.x) >> 5;
  const int m0    = gwave * 16;         // first GEMM row of this wave's tile

  // A: this lane's row of x, pre-offset by its K-half (contiguous k = 16*hi + s).
  const float* arow = x + (size_t)(m0 + lo) * kP + hi * 16;

  v8f acc[4] = {};   // four 16x16 f32 tiles covering N = 0..63

  for (int half = 0; half < 2; ++half) {
    __syncthreads();   // protect LDS reuse across halves (no-op cost on half 0)
    // ---- Stage 512 K-values of KT into LDS, pre-swizzled per lane.
#pragma unroll
    for (int i = 0; i < 8; ++i) {
      const int q   = (int)threadIdx.x + i * 256;  // chunk id 0..2047
      const int kb  = q >> 7;                      // k-block within half
      const int j   = (q >> 5) & 3;                // 16-column tile
      const int l   = q & 31;                      // destination lane
      const int col = j * 16 + (l & 15);           // r column
      const int ks  = half * 512 + kb * 32 + (l >> 4) * 16;
      const v8bf* src = (const v8bf*)(KT + (size_t)col * kP + ks);
      v8bf* dst = (v8bf*)(ldsB + (size_t)q * 16);
      dst[0] = src[0];
      dst[1] = src[1];
    }
    __syncthreads();

    // ---- 16 k-steps of 32 over this half
    for (int kb = 0; kb < 16; ++kb) {
      const int kg = half * 512 + kb * 32;
      // A fragment: x is read exactly once (256 MB) -> non-temporal.
      v4f f0 = __builtin_nontemporal_load((const v4f*)(arow + kg));
      v4f f1 = __builtin_nontemporal_load((const v4f*)(arow + kg + 4));
      v4f f2 = __builtin_nontemporal_load((const v4f*)(arow + kg + 8));
      v4f f3 = __builtin_nontemporal_load((const v4f*)(arow + kg + 12));
      v16bf afrag;
#pragma unroll
      for (int i = 0; i < 4; ++i) {
        afrag[i]      = (__bf16)f0[i];
        afrag[4 + i]  = (__bf16)f1[i];
        afrag[8 + i]  = (__bf16)f2[i];
        afrag[12 + i] = (__bf16)f3[i];
      }

      const __bf16* bp = ldsB + (size_t)((kb * 4) * 32 + lane) * 16;
#pragma unroll
      for (int j = 0; j < 4; ++j) {
        const v8bf* bsrc = (const v8bf*)(bp + (size_t)j * 32 * 16);
        v8bf b0 = bsrc[0];
        v8bf b1 = bsrc[1];
        v16bf bfrag;
#pragma unroll
        for (int i = 0; i < 8; ++i) { bfrag[i] = b0[i]; bfrag[8 + i] = b1[i]; }
        acc[j] = __builtin_amdgcn_wmma_f32_16x16x32_bf16(
            /*neg_a=*/false, afrag, /*neg_b=*/false, bfrag,
            /*c_mod=*/(short)0, acc[j], /*reuse_a=*/false, /*reuse_b=*/false);
      }
    }
  }

  // ---- Fused c-reduction epilogue: the 16 rows of this tile are consecutive
  // c's of one batch element b (C=512 is a multiple of 16).
  const int bidx  = m0 >> 9;            // m0 / 512
  const int cbase = m0 & 511;           // m0 % 512
  float* t3row = t3 + bidx * kR;
#pragma unroll
  for (int j = 0; j < 4; ++j) {
    const int col = j * 16 + lo;        // r index (C/D layout: N = lane%16)
    float partial = 0.0f;
#pragma unroll
    for (int v = 0; v < 8; ++v) {       // C/D layout: M = v + 8*hi
      const int c = cbase + v + hi * 8;
      partial += acc[j][v] * U3[c * kR + col];
    }
    atomicAdd(&t3row[col], partial);    // 64 adds per address total
  }
}

// ---------------------------------------------------------------------------
// Kernel 2: out[b,cls] = sum_r t3[b,r] * lam[r] * U4[cls,r]   (16 MFLOP)
// ---------------------------------------------------------------------------
__global__ void __launch_bounds__(256)
cp_classify(const float* __restrict__ t3, const float* __restrict__ lam,
            const float* __restrict__ U4, float* __restrict__ out) {
  int gid = blockIdx.x * 256 + threadIdx.x;
  if (gid >= kB * kCLS) return;
  int bidx = gid / kCLS;
  int cls  = gid - bidx * kCLS;
  const float* tr = t3 + bidx * kR;
  const float* ur = U4 + (size_t)cls * kR;
  float s = 0.0f;
#pragma unroll
  for (int r = 0; r < kR; ++r) s += tr[r] * lam[r] * ur[r];
  out[gid] = s;
}

// ---------------------------------------------------------------------------
extern "C" void kernel_launch(void* const* d_in, const int* in_sizes, int n_in,
                              void* d_out, int out_size, void* d_ws, size_t ws_size,
                              hipStream_t stream) {
  const float* x   = (const float*)d_in[0];   // (B, C, W, H) f32
  const float* U1  = (const float*)d_in[1];   // (H, R)
  const float* U2  = (const float*)d_in[2];   // (W, R)
  const float* U3  = (const float*)d_in[3];   // (C, R)
  const float* U4  = (const float*)d_in[4];   // (CLS, R)
  const float* lam = (const float*)d_in[5];   // (R,)
  float* out = (float*)d_out;                 // (B, CLS) f32

  // Workspace: KT bf16 [R][P] = 128 KB, then t3 f32 [B][R] = 32 KB.
  __bf16* KT = (__bf16*)d_ws;
  float*  t3 = (float*)((char*)d_ws + (size_t)kR * kP * sizeof(__bf16));

  cp_build_kt<<<(kR * kP) / 256, 256, 0, stream>>>(U1, U2, KT, t3);
  cp_contract<<<kM / 128, 256, 0, stream>>>(x, KT, U3, t3);   // 512 blocks
  cp_classify<<<(kB * kCLS + 255) / 256, 256, 0, stream>>>(t3, lam, U4, out);
}
